// Upsample_52561809769004
// MI455X (gfx1250) — compile-verified, matrix operand order checked by reading, not run on hardware
//
#include <hip/hip_runtime.h>
#include <hip/hip_bf16.h>

// ---------------------------------------------------------------------------
// Deformable conv + BN/ReLU + transposed conv + BN/ReLU for MI455X (gfx1250).
// Big contractions on v_wmma_f32_16x16x32_bf16; double-buffered LDS tiles
// filled with async global->LDS (ASYNCcnt) copies overlapped with compute.
// ---------------------------------------------------------------------------

typedef __bf16 v16bf __attribute__((ext_vector_type(16)));
typedef float  v8f   __attribute__((ext_vector_type(8)));

#define Bsz   8
#define CIN   256
#define COUT  128
#define HH    64
#define WW    64
#define HWSZ  4096          // 64*64
#define KK1   2304          // CIN*9
#define KK2   512           // COUT*4

__device__ __forceinline__ unsigned short f2bf(float f) {
    unsigned int u = __float_as_uint(f);
    unsigned int r = u + 0x7FFFu + ((u >> 16) & 1u);   // round-to-nearest-even
    return (unsigned short)(r >> 16);
}
__device__ __forceinline__ float samp(const float* __restrict__ xc, int y, int x) {
    return (y >= 0 && y < HH && x >= 0 && x < WW) ? xc[(y << 6) + x] : 0.0f;
}

// Async global->LDS 16B copy (GLOBAL_LOAD_ASYNC_TO_LDS_B128, tracked ASYNCcnt).
// LDS byte offset = low 32 bits of the flat shared address (ISA 10.2 aperture map).
__device__ __forceinline__ void async_copy_b128(const void* g, void* l) {
    unsigned lds_off = (unsigned)(uintptr_t)l;
    asm volatile("global_load_async_to_lds_b128 %0, %1, off"
                 :: "v"(lds_off), "v"(g) : "memory");
}
__device__ __forceinline__ void wait_async_zero() {
    asm volatile("s_wait_asynccnt 0x0" ::: "memory");
}

// ---------------------------------------------------------------------------
// 1) Offset head: 3x3 conv 256->27 (+bias), NCHW out: offs[b][oc][h][w]
// ---------------------------------------------------------------------------
__global__ __launch_bounds__(256) void conv_off_kernel(
    const float* __restrict__ x, const float* __restrict__ w_off,
    const float* __restrict__ b_off, float* __restrict__ offs)
{
    __shared__ float lw[8 * 27 * 9];          // 8-channel weight chunk
    const int tid = threadIdx.x;
    const int gid = blockIdx.x * 256 + tid;   // 32768 = B*H*W
    const int b = gid >> 12;
    const int m = gid & 4095;
    const int h = m >> 6, w = m & 63;
    float acc[27];
#pragma unroll
    for (int i = 0; i < 27; ++i) acc[i] = 0.0f;
    const float* xb = x + (size_t)b * CIN * HWSZ;

    for (int c0 = 0; c0 < CIN; c0 += 8) {
        for (int i = tid; i < 8 * 27 * 9; i += 256) {
            int cl = i / (27 * 9), r = i % (27 * 9), oc = r / 9, t = r % 9;
            lw[i] = w_off[((oc * CIN) + (c0 + cl)) * 9 + t];
        }
        __syncthreads();
        for (int cl = 0; cl < 8; ++cl) {
            const float* xc = xb + (size_t)(c0 + cl) * HWSZ;
            float xv[9];
#pragma unroll
            for (int t = 0; t < 9; ++t)
                xv[t] = samp(xc, h + t / 3 - 1, w + t % 3 - 1);
            const float* lwc = &lw[cl * 27 * 9];
#pragma unroll
            for (int oc = 0; oc < 27; ++oc) {
                float s = 0.0f;
#pragma unroll
                for (int t = 0; t < 9; ++t) s += xv[t] * lwc[oc * 9 + t];
                acc[oc] += s;
            }
        }
        __syncthreads();
    }
#pragma unroll
    for (int oc = 0; oc < 27; ++oc)
        offs[(((size_t)b * 27 + oc) << 12) + m] = acc[oc] + b_off[oc];
}

// ---------------------------------------------------------------------------
// 2) Deformable bilinear sampling -> bf16 im2col A1[m][c*9+t] for one batch
// ---------------------------------------------------------------------------
__global__ __launch_bounds__(256) void sample_im2col_kernel(
    const float* __restrict__ x, const float* __restrict__ offs,
    unsigned short* __restrict__ A1, int batch)
{
    const int gid = blockIdx.x * 256 + threadIdx.x;   // 4096*256 threads
    const int c = gid & 255;
    const int m = gid >> 8;
    const int h = m >> 6, w = m & 63;
    const float* xc = x + (((size_t)batch * CIN + c) << 12);
    const float* ob = offs + ((size_t)batch * 27 << 12);
    unsigned short* arow = A1 + (size_t)m * KK1 + c * 9;
#pragma unroll
    for (int t = 0; t < 9; ++t) {
        float oy = ob[((size_t)t << 12) + m];
        float ox = ob[((size_t)(9 + t) << 12) + m];
        float mk = ob[((size_t)(18 + t) << 12) + m];
        float mask = 1.0f / (1.0f + __expf(-mk));
        float py = oy + (float)(h - 1 + t / 3);
        float px = ox + (float)(w - 1 + t % 3);
        float y0f = floorf(py), x0f = floorf(px);
        float ly = py - y0f, lx = px - x0f;
        int y0 = (int)y0f, x0 = (int)x0f;
        float v = (1.0f - ly) * (1.0f - lx) * samp(xc, y0, x0)
                + (1.0f - ly) * lx          * samp(xc, y0, x0 + 1)
                + ly * (1.0f - lx)          * samp(xc, y0 + 1, x0)
                + ly * lx                   * samp(xc, y0 + 1, x0 + 1);
        arow[t] = f2bf(v * mask);
    }
}

// ---------------------------------------------------------------------------
// 3) Weight preps -> bf16, N-MAJOR (row n, K contiguous) for wide LDS loads
// ---------------------------------------------------------------------------
__global__ __launch_bounds__(256) void wdcn_prep_kernel(
    const float* __restrict__ w, unsigned short* __restrict__ B1t)
{
    int gid = blockIdx.x * 256 + threadIdx.x;         // 128*2304 (native [n][k])
    B1t[gid] = f2bf(w[gid]);
}

__global__ __launch_bounds__(256) void wct_prep_kernel(
    const float* __restrict__ w, unsigned short* __restrict__ B2t)
{
    int gid = blockIdx.x * 256 + threadIdx.x;         // 4*128*512
    int cls = gid >> 16;
    int rem = gid & 65535;
    int o = rem >> 9, k = rem & 511;                  // B2t[cls][o][k]
    int c = k >> 2, tap = k & 3;
    int dy = tap >> 1, dx = tap & 1;
    int pY = cls >> 1, pX = cls & 1;
    int wr = pY ? (2 - 2 * dy) : (3 - 2 * dy);        // transposed-conv tap map
    int wc = pX ? (2 - 2 * dx) : (3 - 2 * dx);
    B2t[gid] = f2bf(w[(((size_t)c * COUT + o) * 4 + wr) * 4 + wc]);
}

// ---------------------------------------------------------------------------
// 4) WMMA GEMM: C[4096 x 128] = A[4096 x K] * B[K x 128], bf16 in, f32 acc.
//    A: M-major [m][k]; B passed N-major [n][k] (both K contiguous).
//    Block tile 128x128; 8 waves, each wave = 32x64 (2 M x 4 N WMMA tiles).
//    Double-buffered LDS; async global->LDS copy of tile t+1 overlaps
//    the WMMA work on tile t.
//    mode 0: NCHW store + bias (stage 1). mode 1: parity scatter (stage 2).
// ---------------------------------------------------------------------------
__global__ __launch_bounds__(256) void gemm_bf16_kernel(
    const unsigned short* __restrict__ A, const unsigned short* __restrict__ Bt,
    const float* __restrict__ bias, float* __restrict__ out,
    int K, int batch, int mode, int pY, int pX)
{
    __shared__ unsigned short ldsA[2][128 * 32];   // 2 x 8 KB
    __shared__ unsigned short ldsB[2][128 * 32];   // 2 x 8 KB
    const int tid  = threadIdx.x;
    const int m0   = blockIdx.x * 128;
    const int lane = tid & 31;
    const int wid  = tid >> 5;
    const int mt   = wid & 3;        // 32-row M group
    const int nh   = wid >> 2;       // 64-col N half
    const int r    = lane & 15;
    const int hf   = lane >> 4;      // ISA 16-bit layout: K base = hf*8

    union U { uint4 q[2]; v16bf v; };

    v8f acc[2][4];
#pragma unroll
    for (int mi = 0; mi < 2; ++mi)
#pragma unroll
        for (int j = 0; j < 4; ++j)
#pragma unroll
            for (int v = 0; v < 8; ++v) acc[mi][j][v] = 0.0f;

    const int srow = tid >> 2;             // staging row (0..63), +64 on it 1
    const int scol = (tid & 3) << 3;       // staging col (0,8,16,24)

    auto stage = [&](int buf, int kof) {   // 4 async b128 per thread
#pragma unroll
        for (int it = 0; it < 2; ++it) {
            int row = srow + it * 64;
            async_copy_b128(A + (size_t)(m0 + row) * K + kof + scol,
                            &ldsA[buf][row * 32 + scol]);
            async_copy_b128(Bt + (size_t)row * K + kof + scol,
                            &ldsB[buf][row * 32 + scol]);
        }
    };

    stage(0, 0);
    wait_async_zero();
    __syncthreads();

    for (int kk = 0; kk < K; kk += 32) {
        const int cur = (kk >> 5) & 1;
        if (kk + 32 < K) stage(cur ^ 1, kk + 32);          // overlap next tile
        if (kk + 64 < K)                                    // global_prefetch_b8
            __builtin_prefetch(A + (size_t)(m0 + srow) * K + kk + 64 + scol, 0, 3);

        // 4 B fragments (reused by both M tiles), then 2 A fragments.
        U bu[4];
#pragma unroll
        for (int j = 0; j < 4; ++j) {
            const uint4* pb = (const uint4*)&ldsB[cur][(nh * 64 + j * 16 + r) * 32];
            bu[j].q[0] = pb[hf];
            bu[j].q[1] = pb[2 + hf];
        }
#pragma unroll
        for (int mi = 0; mi < 2; ++mi) {
            U au;
            const uint4* pa = (const uint4*)&ldsA[cur][(mt * 32 + mi * 16 + r) * 32];
            au.q[0] = pa[hf];
            au.q[1] = pa[2 + hf];
#pragma unroll
            for (int j = 0; j < 4; ++j)
                acc[mi][j] = __builtin_amdgcn_wmma_f32_16x16x32_bf16(
                    false, au.v, false, bu[j].v, (short)0, acc[mi][j], false, false);
        }
        wait_async_zero();   // next tile landed (per wave), then block-sync
        __syncthreads();
    }

    // Epilogue. D layout: lane (lane&15)=col; VGPR v -> row v + 8*(lane>>4)
#pragma unroll
    for (int mi = 0; mi < 2; ++mi)
#pragma unroll
        for (int j = 0; j < 4; ++j) {
            int n = nh * 64 + j * 16 + r;
            float bv = (mode == 0) ? bias[n] : 0.0f;
#pragma unroll
            for (int v = 0; v < 8; ++v) {
                int m = m0 + mt * 32 + mi * 16 + (hf << 3) + v;
                float val = acc[mi][j][v] + bv;
                if (mode == 0) {
                    out[(((size_t)batch * COUT + n) << 12) + m] = val;
                } else {
                    int u = m >> 6, w2 = m & 63;
                    int Y = (u << 1) + pY, X = (w2 << 1) + pX;
                    out[(((size_t)batch * COUT + n) * 128 + Y) * 128 + X] = val;
                }
            }
        }
}

// ---------------------------------------------------------------------------
// 5) BatchNorm stats (population mean/var per channel), LDS tree reduction
// ---------------------------------------------------------------------------
__global__ __launch_bounds__(256) void bn_stats_kernel(
    const float* __restrict__ src, float* __restrict__ mean,
    float* __restrict__ rstd, int spatial)
{
    __shared__ float ss[256], ss2[256];
    const int c = blockIdx.x, tid = threadIdx.x;
    float s = 0.0f, s2 = 0.0f;
    for (int b = 0; b < Bsz; ++b) {
        const float* p = src + ((size_t)b * COUT + c) * (size_t)spatial;
        for (int i = tid; i < spatial; i += 256) {
            float v = p[i]; s += v; s2 += v * v;
        }
    }
    ss[tid] = s; ss2[tid] = s2;
    __syncthreads();
    for (int o = 128; o > 0; o >>= 1) {
        if (tid < o) { ss[tid] += ss[tid + o]; ss2[tid] += ss2[tid + o]; }
        __syncthreads();
    }
    if (tid == 0) {
        float N = (float)Bsz * (float)spatial;
        float mu = ss[0] / N;
        float var = ss2[0] / N - mu * mu;
        mean[c] = mu;
        rstd[c] = rsqrtf(var + 1e-5f);
    }
}

__global__ __launch_bounds__(256) void bn_relu_bf16_kernel(
    const float* __restrict__ src, unsigned short* __restrict__ dst,
    const float* __restrict__ mean, const float* __restrict__ rstd,
    const float* __restrict__ g, const float* __restrict__ be)
{
    size_t idx = (size_t)blockIdx.x * 256 + threadIdx.x;   // 4194304 elems
    int c = (int)((idx >> 12) & 127);
    float v = (src[idx] - mean[c]) * rstd[c] * g[c] + be[c];
    dst[idx] = f2bf(v > 0.0f ? v : 0.0f);
}

__global__ __launch_bounds__(256) void bn_relu_f32_kernel(
    float* __restrict__ data,
    const float* __restrict__ mean, const float* __restrict__ rstd,
    const float* __restrict__ g, const float* __restrict__ be)
{
    size_t idx = (size_t)blockIdx.x * 256 + threadIdx.x;   // 16777216 elems
    int c = (int)((idx >> 14) & 127);
    float v = (data[idx] - mean[c]) * rstd[c] * g[c] + be[c];
    data[idx] = v > 0.0f ? v : 0.0f;
}

// ---------------------------------------------------------------------------
// 6) Stage-2 im2col: 2x2-tap parity-class patches from bf16 stage-1 output
// ---------------------------------------------------------------------------
__global__ __launch_bounds__(256) void im2col2_kernel(
    const unsigned short* __restrict__ out1bf, unsigned short* __restrict__ A2,
    int batch, int pY, int pX)
{
    const int gid = blockIdx.x * 256 + threadIdx.x;   // 4096*128 threads
    const int c = gid & 127;
    const int m = gid >> 7;
    const int u = m >> 6, vv = m & 63;
    const unsigned short* src = out1bf + (((size_t)batch * COUT + c) << 12);
    unsigned short* dst = A2 + (size_t)m * KK2 + (c << 2);
#pragma unroll
    for (int dy = 0; dy < 2; ++dy)
#pragma unroll
        for (int dx = 0; dx < 2; ++dx) {
            int y = pY ? (u + dy) : (u - 1 + dy);
            int x = pX ? (vv + dx) : (vv - 1 + dx);
            unsigned short val = 0;   // bf16 zero
            if (y >= 0 && y < HH && x >= 0 && x < WW) val = src[(y << 6) + x];
            dst[dy * 2 + dx] = val;
        }
}

// ---------------------------------------------------------------------------
// Host launcher
// ---------------------------------------------------------------------------
extern "C" void kernel_launch(void* const* d_in, const int* in_sizes, int n_in,
                              void* d_out, int out_size, void* d_ws, size_t ws_size,
                              hipStream_t stream)
{
    (void)in_sizes; (void)n_in; (void)out_size; (void)ws_size;
    const float* x      = (const float*)d_in[0];
    const float* w_off  = (const float*)d_in[1];
    const float* b_off  = (const float*)d_in[2];
    const float* w_dcn  = (const float*)d_in[3];
    const float* b_dcn  = (const float*)d_in[4];
    const float* gamma1 = (const float*)d_in[5];
    const float* beta1  = (const float*)d_in[6];
    const float* w_ct   = (const float*)d_in[7];
    const float* gamma2 = (const float*)d_in[8];
    const float* beta2  = (const float*)d_in[9];
    float* out = (float*)d_out;

    char* base = (char*)d_ws;
    size_t off = 0;
    auto carve = [&](size_t bytes) -> char* {
        char* p = base + off;
        off = (off + bytes + 255) & ~(size_t)255;
        return p;
    };
    float*          offs   = (float*)carve((size_t)Bsz * 27 * HWSZ * 4);   // 3.5 MB
    float*          out1   = (float*)carve((size_t)Bsz * COUT * HWSZ * 4); // 16.8 MB
    unsigned short* out1bf = (unsigned short*)carve((size_t)Bsz * COUT * HWSZ * 2);
    unsigned short* A1     = (unsigned short*)carve((size_t)HWSZ * KK1 * 2); // 18.9 MB
    unsigned short* B1t    = (unsigned short*)carve((size_t)COUT * KK1 * 2);
    unsigned short* A2     = (unsigned short*)carve((size_t)HWSZ * KK2 * 2); // 4.2 MB
    unsigned short* B2t    = (unsigned short*)carve((size_t)4 * COUT * KK2 * 2);
    float* mean1 = (float*)carve(128 * 4);
    float* rstd1 = (float*)carve(128 * 4);
    float* mean2 = (float*)carve(128 * 4);
    float* rstd2 = (float*)carve(128 * 4);

    // Stage 0: offset head + weight preps
    conv_off_kernel<<<128, 256, 0, stream>>>(x, w_off, b_off, offs);
    wdcn_prep_kernel<<<1152, 256, 0, stream>>>(w_dcn, B1t);
    wct_prep_kernel<<<1024, 256, 0, stream>>>(w_ct, B2t);

    // Stage 1: deformable sampling + WMMA GEMM per batch
    for (int b = 0; b < Bsz; ++b) {
        sample_im2col_kernel<<<4096, 256, 0, stream>>>(x, offs, A1, b);
        gemm_bf16_kernel<<<32, 256, 0, stream>>>(A1, B1t, b_dcn, out1,
                                                 KK1, b, 0, 0, 0);
    }
    bn_stats_kernel<<<128, 256, 0, stream>>>(out1, mean1, rstd1, HWSZ);
    bn_relu_bf16_kernel<<<16384, 256, 0, stream>>>(out1, out1bf, mean1, rstd1,
                                                   gamma1, beta1);

    // Stage 2: transposed conv as 4 parity-class WMMA GEMMs per batch
    for (int b = 0; b < Bsz; ++b)
        for (int cls = 0; cls < 4; ++cls) {
            int pY = cls >> 1, pX = cls & 1;
            im2col2_kernel<<<2048, 256, 0, stream>>>(out1bf, A2, b, pY, pX);
            gemm_bf16_kernel<<<32, 256, 0, stream>>>(A2, B2t + (size_t)cls * COUT * KK2,
                                                     b_dcn /*unused*/, out,
                                                     KK2, b, 1, pY, pX);
        }

    // Final BN + ReLU in place on d_out
    bn_stats_kernel<<<128, 256, 0, stream>>>(out, mean2, rstd2, 128 * 128);
    bn_relu_f32_kernel<<<65536, 256, 0, stream>>>(out, mean2, rstd2, gamma2, beta2);
}
